// MultiheadAttention_90383291777201
// MI455X (gfx1250) — compile-verified, matrix operand order checked by reading, not run on hardware
//
#include <hip/hip_runtime.h>

// ---------------------------------------------------------------------------
// MultiheadAttention (argmax variant) for MI455X / gfx1250, wave32 + WMMA.
//   B=8, S=2048, D=768.
//   q = X Wq^T + bq ; k = ...; v = ...            (bf16 WMMA, f32 accumulate)
//   scores = q k^T, masked, argmax over keys      (bf16 WMMA, f32 accumulate)
//   out0 = v[idx] / sqrt(D) ; out1 = one_hot(idx)
// ---------------------------------------------------------------------------

#define B_DIM 8
#define S_DIM 2048
#define D_DIM 768
#define M_DIM (B_DIM * S_DIM)          // 16384 rows for the projection GEMMs
#define KT_STEPS (D_DIM / 32)          // 24 K-steps of 32

typedef __bf16          bf16x8  __attribute__((ext_vector_type(8)));
typedef __bf16          bf16x16 __attribute__((ext_vector_type(16)));
typedef float           f32x8   __attribute__((ext_vector_type(8)));
typedef unsigned short  u16x8   __attribute__((ext_vector_type(8)));
typedef unsigned short  u16x16  __attribute__((ext_vector_type(16)));

// float -> bf16 via the hardware convert (RNE on gfx1250)
__device__ __forceinline__ unsigned short f2bf(float f) {
  __bf16 h = (__bf16)f;
  return __builtin_bit_cast(unsigned short, h);
}

// Pack 8 floats into 8 bf16 using hardware converts.
__device__ __forceinline__ u16x8 pack8(float4 a, float4 b) {
  bf16x8 r = {(__bf16)a.x, (__bf16)a.y, (__bf16)a.z, (__bf16)a.w,
              (__bf16)b.x, (__bf16)b.y, (__bf16)b.z, (__bf16)b.w};
  return __builtin_bit_cast(u16x8, r);
}

// Build a 16x32-bf16 A/B fragment for v_wmma_f32_16x16x32_bf16.
// ISA layout (16-bit A, 16x32): lane L holds row (L&15); lanes 0-15 use
// k-offset 0, lanes 16-31 use k-offset 8.  VGPR0..3 = k..k+7 (packed pairs),
// VGPR4..7 = k+16..k+23.  So each lane loads p[0..7] and p[16..23] where
// p = &row[khalf*8].  The "N-major" B tile (key/weight rows stored row-major)
// has the identical pattern with row = N column -> two b128 loads, no shuffle.
__device__ __forceinline__ bf16x16 frag_ld(const unsigned short* p) {
  u16x8 lo = *(const u16x8*)p;
  u16x8 hi = *(const u16x8*)(p + 16);
  u16x16 t;
#pragma unroll
  for (int i = 0; i < 8; ++i) { t[i] = lo[i]; t[i + 8] = hi[i]; }
  return __builtin_bit_cast(bf16x16, t);
}

// ---------------------------------------------------------------------------
// Kernel 1: C[M,768] = X[M,768] * W[768,768]^T + bias.  W row-major [o][d],
// so the WMMA B tile (K x N with N = o) is exactly W rows -> contiguous loads.
// Block tile 128x64, 256 threads (8 waves); each wave computes a 32x32 patch
// (4 WMMAs / K-step, 4 independent accumulator chains).  LDS panels are
// double-buffered: next panel's global loads issue before this panel's WMMAs,
// one barrier per K-step.  Writes bf16 and/or f32 outputs (nullable ptrs).
// ---------------------------------------------------------------------------
__global__ __launch_bounds__(256) void proj_gemm_kernel(
    const float* __restrict__ X, const float* __restrict__ W,
    const float* __restrict__ bias,
    unsigned short* __restrict__ out_bf, float* __restrict__ out_f) {
  __shared__ __align__(16) unsigned short As[2][128][40];  // 20 KB (+pad)
  __shared__ __align__(16) unsigned short Bs[2][64][40];   // 10 KB

  const int m0 = blockIdx.x * 128;
  const int n0 = blockIdx.y * 64;
  const int tid = threadIdx.x;
  const int lane = tid & 31;
  const int w = tid >> 5;
  const int wm = (w >> 1) * 32;        // wave's M offset inside block tile
  const int wn = (w & 1) * 32;         // wave's N offset
  const int lm = lane & 15;
  const int kh = lane >> 4;

  // Staging map: A panel 128x32 -> 16 floats/thread, B panel 64x32 -> 8.
  const int arow = tid >> 1, acol = (tid & 1) * 16;
  const int brow = tid >> 2, bcol = (tid & 3) * 8;
  const float* aptr = X + (size_t)(m0 + arow) * D_DIM + acol;
  const float* bptr = W + (size_t)(n0 + brow) * D_DIM + bcol;

  const f32x8 zero = {0, 0, 0, 0, 0, 0, 0, 0};
  f32x8 acc00 = zero, acc01 = zero, acc10 = zero, acc11 = zero;

  float4 xa, xb, xc, xd, wa, wb;
  // Preload K-step 0 into registers, stage into buffer 0.
  xa = *(const float4*)(aptr + 0);  xb = *(const float4*)(aptr + 4);
  xc = *(const float4*)(aptr + 8);  xd = *(const float4*)(aptr + 12);
  wa = *(const float4*)(bptr + 0);  wb = *(const float4*)(bptr + 4);
  *(u16x8*)&As[0][arow][acol]     = pack8(xa, xb);
  *(u16x8*)&As[0][arow][acol + 8] = pack8(xc, xd);
  *(u16x8*)&Bs[0][brow][bcol]     = pack8(wa, wb);

  for (int kt = 0; kt < KT_STEPS; ++kt) {
    __syncthreads();  // buffer (kt&1) visible to every wave
    const int cur = kt & 1;
    if (kt + 1 < KT_STEPS) {  // issue next panel's global loads early
      const int k1 = (kt + 1) * 32;
      xa = *(const float4*)(aptr + k1 + 0);  xb = *(const float4*)(aptr + k1 + 4);
      xc = *(const float4*)(aptr + k1 + 8);  xd = *(const float4*)(aptr + k1 + 12);
      wa = *(const float4*)(bptr + k1 + 0);  wb = *(const float4*)(bptr + k1 + 4);
    }
    bf16x16 a0 = frag_ld(&As[cur][wm + lm][kh * 8]);
    bf16x16 a1 = frag_ld(&As[cur][wm + 16 + lm][kh * 8]);
    bf16x16 b0 = frag_ld(&Bs[cur][wn + lm][kh * 8]);
    bf16x16 b1 = frag_ld(&Bs[cur][wn + 16 + lm][kh * 8]);
    acc00 = __builtin_amdgcn_wmma_f32_16x16x32_bf16(
        false, a0, false, b0, (short)0, acc00, false, false);
    acc01 = __builtin_amdgcn_wmma_f32_16x16x32_bf16(
        false, a0, false, b1, (short)0, acc01, false, false);
    acc10 = __builtin_amdgcn_wmma_f32_16x16x32_bf16(
        false, a1, false, b0, (short)0, acc10, false, false);
    acc11 = __builtin_amdgcn_wmma_f32_16x16x32_bf16(
        false, a1, false, b1, (short)0, acc11, false, false);
    if (kt + 1 < KT_STEPS) {  // stage next panel into the other buffer
      const int nxt = cur ^ 1;
      *(u16x8*)&As[nxt][arow][acol]     = pack8(xa, xb);
      *(u16x8*)&As[nxt][arow][acol + 8] = pack8(xc, xd);
      *(u16x8*)&Bs[nxt][brow][bcol]     = pack8(wa, wb);
    }
  }

  // C layout: lanes 0-15 -> N=lane, M=r ; lanes 16-31 -> N=lane-16, M=r+8
  const int ng0 = n0 + wn + lm;
  const int ng1 = ng0 + 16;
  const float bias0 = bias[ng0];
  const float bias1 = bias[ng1];
#pragma unroll
  for (int r = 0; r < 8; ++r) {
    const size_t mg0 = (size_t)(m0 + wm + kh * 8 + r);
    const size_t mg1 = mg0 + 16;
    const float v00 = acc00[r] + bias0, v01 = acc01[r] + bias1;
    const float v10 = acc10[r] + bias0, v11 = acc11[r] + bias1;
    if (out_bf) {
      out_bf[mg0 * D_DIM + ng0] = f2bf(v00);
      out_bf[mg0 * D_DIM + ng1] = f2bf(v01);
      out_bf[mg1 * D_DIM + ng0] = f2bf(v10);
      out_bf[mg1 * D_DIM + ng1] = f2bf(v11);
    }
    if (out_f) {
      out_f[mg0 * D_DIM + ng0] = v00;
      out_f[mg0 * D_DIM + ng1] = v01;
      out_f[mg1 * D_DIM + ng0] = v10;
      out_f[mg1 * D_DIM + ng1] = v11;
    }
  }
}

// ---------------------------------------------------------------------------
// Kernel 2: one block per (batch b, 16-query-row tile).
//   scores[16,2048] = q_tile * K^T via bf16 WMMA (K rows streamed, L2
//   resident: 3 MB/batch), mask folded into a per-column 16-bit word,
//   running argmax per lane, block reduction, then one-hot + V gather.
// ---------------------------------------------------------------------------
__global__ __launch_bounds__(256) void attn_argmax_kernel(
    const unsigned short* __restrict__ qb, const unsigned short* __restrict__ kb,
    const float* __restrict__ vf, const unsigned char* __restrict__ attn_mask,
    const unsigned char* __restrict__ kpm,
    float* __restrict__ out_attn, float* __restrict__ out_w) {
  __shared__ __align__(16) unsigned short Qs[16][D_DIM];  // 24 KB
  __shared__ unsigned int colmask[S_DIM];                 // 8 KB
  __shared__ float Rmax[256][8];                          // 8 KB
  __shared__ int Ridx[256][8];                            // 8 KB
  __shared__ unsigned char rowpad[16];
  __shared__ int RowIdx[16];

  const int b = blockIdx.x >> 7;                // S/16 = 128 row tiles / batch
  const int m0 = (blockIdx.x & 127) * 16;
  const int tid = threadIdx.x;
  const int lane = tid & 31;
  const int w = tid >> 5;
  const int lm = lane & 15;
  const int kh = lane >> 4;

  // Stage the 16x768 bf16 query tile.
  const unsigned short* qsrc = qb + ((size_t)b * S_DIM + m0) * D_DIM;
  for (int i = tid; i < 16 * D_DIM / 8; i += 256)
    ((uint4*)&Qs[0][0])[i] = ((const uint4*)qsrc)[i];
  if (tid < 16) rowpad[tid] = kpm[(size_t)b * S_DIM + m0 + tid];
  __syncthreads();

  unsigned int rowbits = 0;
#pragma unroll
  for (int r = 0; r < 16; ++r) rowbits |= ((unsigned)(rowpad[r] != 0)) << r;

  // colmask[n] bit m == (attn_mask[m0+m][n] | kpm[b][n] | kpm[b][m0+m])
  for (int n = tid; n < S_DIM; n += 256) {
    unsigned int cm = kpm[(size_t)b * S_DIM + n] ? 0xFFFFu : rowbits;
    unsigned int cb = 0;
    const unsigned char* am = attn_mask + n;
#pragma unroll
    for (int r = 0; r < 16; ++r)
      cb |= ((unsigned)(am[(size_t)(m0 + r) * S_DIM] != 0)) << r;
    colmask[n] = cm | cb;
  }
  __syncthreads();

  float best[8];
  int bidx[8];
#pragma unroll
  for (int r = 0; r < 8; ++r) { best[r] = -__builtin_inff(); bidx[r] = 0; }

  const size_t kbase = (size_t)b * S_DIM * D_DIM;
  const f32x8 zero = {0, 0, 0, 0, 0, 0, 0, 0};

  // Wave w owns key tiles [w*16, w*16+16), processed in groups of 4 so four
  // independent WMMA chains stay in flight and the A fragment is reused 4x.
  for (int g = 0; g < 4; ++g) {
    const int ctb = w * 16 + g * 4;
    f32x8 acc[4] = {zero, zero, zero, zero};
    const unsigned short* krow[4];
#pragma unroll
    for (int j = 0; j < 4; ++j) {
      krow[j] = kb + kbase + (size_t)((ctb + j) * 16 + lm) * D_DIM;
      __builtin_prefetch(krow[j], 0, 1);  // global_prefetch_b8
    }
    for (int kt = 0; kt < KT_STEPS; ++kt) {
      const int ko = kt * 32 + kh * 8;
      bf16x16 af = frag_ld(&Qs[lm][ko]);
#pragma unroll
      for (int j = 0; j < 4; ++j) {
        bf16x16 bfj = frag_ld(krow[j] + ko);
        acc[j] = __builtin_amdgcn_wmma_f32_16x16x32_bf16(
            false, af, false, bfj, (short)0, acc[j], false, false);
      }
    }
#pragma unroll
    for (int j = 0; j < 4; ++j) {
      const int nglob = (ctb + j) * 16 + lm;
      const unsigned int cm = colmask[nglob];
#pragma unroll
      for (int r = 0; r < 8; ++r) {
        const int mrow = kh * 8 + r;  // row inside the 16-row tile (C layout)
        const float s = ((cm >> mrow) & 1u) ? -1e10f : acc[j][r];
        if (s > best[r]) { best[r] = s; bidx[r] = nglob; }  // keep first max
      }
    }
  }

#pragma unroll
  for (int r = 0; r < 8; ++r) { Rmax[tid][r] = best[r]; Ridx[tid][r] = bidx[r]; }
  __syncthreads();

  // Reduce 128 candidates per row; ties -> smallest key index (jnp.argmax).
  if (tid < 16) {
    const int rr = tid & 7;
    const int want_kh = tid >> 3;
    float bb = -__builtin_inff();
    int bi = 0x7fffffff;
    for (int w2 = 0; w2 < 8; ++w2)
      for (int l = 0; l < 16; ++l) {
        const int t = w2 * 32 + want_kh * 16 + l;
        const float v = Rmax[t][rr];
        const int ix = Ridx[t][rr];
        if (v > bb || (v == bb && ix < bi)) { bb = v; bi = ix; }
      }
    RowIdx[tid] = bi;
  }
  __syncthreads();

  // attn_weights: zero the 16x2048 strip, then set the one-hot entries.
  float* aw = out_w + ((size_t)b * S_DIM + m0) * S_DIM;
  {
    const int row = tid >> 4, seg = tid & 15;
    float4 z; z.x = 0.f; z.y = 0.f; z.z = 0.f; z.w = 0.f;
    float4* p = (float4*)(aw + (size_t)row * S_DIM + seg * 128);
#pragma unroll
    for (int i = 0; i < 32; ++i) p[i] = z;
  }
  __syncthreads();
  if (tid < 16) aw[(size_t)tid * S_DIM + RowIdx[tid]] = 1.0f;

  // attn_outputs: gather v[b, idx, :] / sqrt(D)
  const float inv = 0.036084391824351615f;  // 1/sqrt(768)
  {
    const int row = tid >> 4, t = tid & 15;
    const float4* s4 =
        (const float4*)(vf + ((size_t)b * S_DIM + RowIdx[row]) * D_DIM);
    float4* d4 = (float4*)(out_attn + ((size_t)b * S_DIM + m0 + row) * D_DIM);
    for (int i = t; i < D_DIM / 4; i += 16) {
      float4 x = s4[i];
      x.x *= inv; x.y *= inv; x.z *= inv; x.w *= inv;
      d4[i] = x;
    }
  }
}

extern "C" void kernel_launch(void* const* d_in, const int* in_sizes, int n_in,
                              void* d_out, int out_size, void* d_ws,
                              size_t ws_size, hipStream_t stream) {
  (void)in_sizes; (void)n_in; (void)out_size; (void)ws_size;
  const float* query = (const float*)d_in[0];
  const float* key_in = (const float*)d_in[1];
  const float* value = (const float*)d_in[2];
  const unsigned char* attn_mask = (const unsigned char*)d_in[3];
  const unsigned char* kpm = (const unsigned char*)d_in[4];
  const float* wq = (const float*)d_in[5];
  const float* bq = (const float*)d_in[6];
  const float* wk = (const float*)d_in[7];
  const float* bk = (const float*)d_in[8];
  const float* wv = (const float*)d_in[9];
  const float* bv = (const float*)d_in[10];

  // Workspace: q_bf16 (24MB) | k_bf16 (24MB) | v_f32 (48MB)
  unsigned short* qb = (unsigned short*)d_ws;
  unsigned short* kb = qb + (size_t)M_DIM * D_DIM;
  float* vf = (float*)(kb + (size_t)M_DIM * D_DIM);

  float* out_attn = (float*)d_out;                          // [B,S,D]
  float* out_w = out_attn + (size_t)M_DIM * D_DIM;          // [B,S,S]

  dim3 gA(M_DIM / 128, D_DIM / 64);
  proj_gemm_kernel<<<gA, 256, 0, stream>>>(query, wq, bq, qb, nullptr);
  proj_gemm_kernel<<<gA, 256, 0, stream>>>(key_in, wk, bk, kb, nullptr);
  proj_gemm_kernel<<<gA, 256, 0, stream>>>(value, wv, bv, nullptr, vf);
  attn_argmax_kernel<<<B_DIM * (S_DIM / 16), 256, 0, stream>>>(
      qb, kb, vf, attn_mask, kpm, out_attn, out_w);
}
